// model_rel_gru_57861799411871
// MI455X (gfx1250) — compile-verified, matrix operand order checked by reading, not run on hardware
//
#include <hip/hip_runtime.h>

// ---------------- model constants (from reference) ----------------
#define A_   512
#define TP_  8
#define TF_  12
#define Cc   512      // GRU hidden
#define Mm   32       // memory width
#define Nn   128      // memory slots
#define Hh   20       // heads
#define EMBv 16
#define Ll   8        // agents per scene
#define Ss   64       // scenes
#define Dd   544      // C + M (decoder hidden)

typedef __bf16 bf16;
typedef __bf16 v16bf __attribute__((ext_vector_type(16)));
typedef float  v8f   __attribute__((ext_vector_type(8)));

__device__ __forceinline__ bf16 f2bf(float f) {
    union { float f; unsigned u; } x; x.f = f;
    unsigned r = (x.u + 0x7fffu + ((x.u >> 16) & 1u)) >> 16;
    union { unsigned short s; bf16 b; } y; y.s = (unsigned short)r;
    return y.b;
}
__device__ __forceinline__ unsigned short f2bfu(float f) {
    union { float f; unsigned u; } x; x.f = f;
    return (unsigned short)((x.u + 0x7fffu + ((x.u >> 16) & 1u)) >> 16);
}
__device__ __forceinline__ float sigmf_(float x) { return 1.f / (1.f + __expf(-x)); }
__device__ __forceinline__ float softplusf_(float x) { return (x > 20.f) ? x : log1pf(__expf(x)); }

// WMMA 16x16x32 bf16 fragment mapping (A 16x32 / mirrored B 32x16):
//   lane half = lane>>4; element i of v16bf:  k = (i<8 ? i : i+8) + 8*half
// Key property: any aligned group of 4 consecutive k (k%8 in {0,4}) lives in
// ONE lane at 4 consecutive element slots -> 8-byte LDS stores for A staging,
// and pre-packed weights give 32B/lane contiguous B fragments in global.

// ---------------------------------------------------------------------------
// Weight pre-pack: W fp32 (Ktot x ngates*gateW, row-major) -> bf16 fragments
// layout [kcI][g][ct][lane][i], frag = 512 bf16 = 1KB, CTpad col-tiles of 16.
// Out-of-range (k,col) -> 0.
// ---------------------------------------------------------------------------
__global__ void pack_wmma_b_kernel(const float* __restrict__ W, bf16* __restrict__ out,
                                   int Ktot, int gateW, int ngates, int CTpad)
{
    int nkc = (Ktot + 31) >> 5;
    size_t total = (size_t)nkc * ngates * CTpad * 512;
    size_t stride = (size_t)gridDim.x * blockDim.x;
    int ld = ngates * gateW;
    for (size_t id = blockIdx.x * (size_t)blockDim.x + threadIdx.x; id < total; id += stride) {
        int i    = (int)(id & 15);
        int lane = (int)((id >> 4) & 31);
        size_t rest = id >> 9;
        int ct = (int)(rest % CTpad); rest /= CTpad;
        int g  = (int)(rest % ngates); rest /= ngates;
        int kcI = (int)rest;
        int k   = kcI * 32 + ((i < 8) ? i : i + 8) + 8 * (lane >> 4);
        int col = ct * 16 + (lane & 15);
        float v = 0.f;
        if (k < Ktot && col < gateW) v = W[(size_t)k * ld + g * gateW + col];
        out[id] = f2bf(v);
    }
}

// fcr_w (H,C,33): logical B is (C x H*33) with col j=h*33+r -> packed fragments
__global__ void pack_fcr_kernel(const float* __restrict__ fcr, bf16* __restrict__ out, int CTpad)
{
    const int Ktot = Cc, NW = Hh * 33;
    int nkc = Ktot >> 5;
    size_t total = (size_t)nkc * CTpad * 512;
    size_t stride = (size_t)gridDim.x * blockDim.x;
    for (size_t id = blockIdx.x * (size_t)blockDim.x + threadIdx.x; id < total; id += stride) {
        int i    = (int)(id & 15);
        int lane = (int)((id >> 4) & 31);
        size_t rest = id >> 9;
        int ct = (int)(rest % CTpad);
        int kcI = (int)(rest / CTpad);
        int k   = kcI * 32 + ((i < 8) ? i : i + 8) + 8 * (lane >> 4);
        int col = ct * 16 + (lane & 15);
        float v = 0.f;
        if (col < NW) {
            int h = col / 33, r = col % 33;
            v = fcr[((size_t)h * Cc + k) * 33 + r];
        }
        out[id] = f2bf(v);
    }
}

// ---------------------------------------------------------------------------
// Fused GRU: h_out = GRU(x, h_in).  Block = 32 rows x 64 gate-cols, 8 waves.
// Six f32 WMMA accumulators per wave; B fragments read directly from packed
// global weights (coalesced b128); A staged via LDS (float4 -> 4x bf16).
// ---------------------------------------------------------------------------
__launch_bounds__(256)
__global__ void gru_fused_wmma(const float* __restrict__ x, int Kx,
                               const float* __restrict__ h_in,
                               float* __restrict__ h_out,
                               const bf16* __restrict__ Wih,   // packed frags
                               const bf16* __restrict__ Whh,   // packed frags
                               const float* __restrict__ bih,
                               const float* __restrict__ bhh,
                               int Mtot, int C, int CTpadIH, int CTpadHH)
{
    __shared__ __align__(32) bf16 ldsA[2][32][16];

    const int tid  = threadIdx.x;
    const int lane = tid & 31;
    const int wv   = tid >> 5;
    const int wr   = wv >> 2;                 // wave row (0..1)
    const int wc   = wv & 3;                  // wave col (0..3)
    const int rowBase = blockIdx.y * 32;      // Mtot is a multiple of 32
    const int ct = blockIdx.x * 4 + wc;       // 16-col tile index

    // A staging slot for this thread: row sm, 4 consecutive k at k4
    const int sm = tid >> 3;
    const int k4 = (tid & 7) * 4;
    const int sHalf = (k4 >> 3) & 1;
    const int sRem  = k4 - 8 * sHalf;
    bf16* sDst = &ldsA[sm >> 4][(sm & 15) + 16 * sHalf][(sRem & 7) + (sRem >= 16 ? 8 : 0)];
    const size_t aRow = (size_t)(rowBase + sm);

    v8f accX[3] = {};   // ir, iz, in
    v8f accH[3] = {};   // hr, hz, hn

    // ---- phase 1: x @ Wih ----
    const int nkcX = (Kx + 31) >> 5;
    for (int kcI = 0; kcI < nkcX; ++kcI) {
        const int kc = kcI << 5;
        __syncthreads();
        {
            float4 v = make_float4(0.f, 0.f, 0.f, 0.f);
            if (kc + k4 + 4 <= Kx)                       // Kx is a multiple of 16
                v = *(const float4*)&x[aRow * Kx + kc + k4];
            ushort4 p;
            p.x = f2bfu(v.x); p.y = f2bfu(v.y); p.z = f2bfu(v.z); p.w = f2bfu(v.w);
            *(ushort4*)sDst = p;                         // single 8B LDS store
        }
        __syncthreads();
        v16bf af = *(const v16bf*)(&ldsA[wr][lane][0]);
        const bf16* bb = Wih + (((size_t)kcI * 3 * CTpadIH + ct) * 32 + lane) * 16;
        const size_t gs = (size_t)CTpadIH * 512;
        v16bf b0 = *(const v16bf*)(bb);
        v16bf b1 = *(const v16bf*)(bb + gs);
        v16bf b2 = *(const v16bf*)(bb + 2 * gs);
        accX[0] = __builtin_amdgcn_wmma_f32_16x16x32_bf16(false, af, false, b0, (short)0, accX[0], false, false);
        accX[1] = __builtin_amdgcn_wmma_f32_16x16x32_bf16(false, af, false, b1, (short)0, accX[1], false, false);
        accX[2] = __builtin_amdgcn_wmma_f32_16x16x32_bf16(false, af, false, b2, (short)0, accX[2], false, false);
    }

    // ---- phase 2: h @ Whh ----
    const int nkcH = C >> 5;                              // C multiple of 32
    for (int kcI = 0; kcI < nkcH; ++kcI) {
        const int kc = kcI << 5;
        __syncthreads();
        {
            float4 v = *(const float4*)&h_in[aRow * C + kc + k4];
            ushort4 p;
            p.x = f2bfu(v.x); p.y = f2bfu(v.y); p.z = f2bfu(v.z); p.w = f2bfu(v.w);
            *(ushort4*)sDst = p;
        }
        const bf16* bb = Whh + (((size_t)kcI * 3 * CTpadHH + ct) * 32 + lane) * 16;
        const size_t gs = (size_t)CTpadHH * 512;
        if (kcI + 1 < nkcH)
            __builtin_prefetch((const void*)(bb + 3 * gs * 0 + (size_t)3 * CTpadHH * 512), 0, 1);
        __syncthreads();
        v16bf af = *(const v16bf*)(&ldsA[wr][lane][0]);
        v16bf b0 = *(const v16bf*)(bb);
        v16bf b1 = *(const v16bf*)(bb + gs);
        v16bf b2 = *(const v16bf*)(bb + 2 * gs);
        accH[0] = __builtin_amdgcn_wmma_f32_16x16x32_bf16(false, af, false, b0, (short)0, accH[0], false, false);
        accH[1] = __builtin_amdgcn_wmma_f32_16x16x32_bf16(false, af, false, b1, (short)0, accH[1], false, false);
        accH[2] = __builtin_amdgcn_wmma_f32_16x16x32_bf16(false, af, false, b2, (short)0, accH[2], false, false);
    }

    // ---- fused GRU gates (C/D layout: vgpr r, lanes 0-15 -> M=r, 16-31 -> M=8+r)
    const int col   = ct * 16 + (lane & 15);
    const int mrow0 = rowBase + wr * 16 + (lane >> 4) * 8;
    if (col < C) {
        float bir = bih[col], biz = bih[C + col], bin_ = bih[2 * C + col];
        float bhr = bhh[col], bhz = bhh[C + col], bhn = bhh[2 * C + col];
#pragma unroll
        for (int r = 0; r < 8; ++r) {
            int m = mrow0 + r;
            float rg = sigmf_(accX[0][r] + bir + accH[0][r] + bhr);
            float zg = sigmf_(accX[1][r] + biz + accH[1][r] + bhz);
            float ng = tanhf(accX[2][r] + bin_ + rg * (accH[2][r] + bhn));
            float hp = h_in[(size_t)m * C + col];
            h_out[(size_t)m * C + col] = (1.f - zg) * ng + zg * hp;
        }
    }
}

// ---------------------------------------------------------------------------
// Generic WMMA GEMM + bias: out[MxN] = A[MxK](fp32) @ Bpacked + bias[N]
// ---------------------------------------------------------------------------
__launch_bounds__(256)
__global__ void gemm_bias_wmma(const float* __restrict__ Amat,
                               const bf16* __restrict__ Bp,
                               const float* __restrict__ bias,
                               float* __restrict__ out,
                               int Mtot, int K, int N, int CTpad)
{
    __shared__ __align__(32) bf16 ldsA[2][32][16];

    const int tid  = threadIdx.x;
    const int lane = tid & 31;
    const int wv   = tid >> 5;
    const int wr   = wv >> 2, wc = wv & 3;
    const int rowBase = blockIdx.y * 32;
    const int ct = blockIdx.x * 4 + wc;

    const int sm = tid >> 3;
    const int k4 = (tid & 7) * 4;
    const int sHalf = (k4 >> 3) & 1;
    const int sRem  = k4 - 8 * sHalf;
    bf16* sDst = &ldsA[sm >> 4][(sm & 15) + 16 * sHalf][(sRem & 7) + (sRem >= 16 ? 8 : 0)];
    const size_t aRow = (size_t)(rowBase + sm);

    v8f acc = {};
    const int nkc = (K + 31) >> 5;
    for (int kcI = 0; kcI < nkc; ++kcI) {
        const int kc = kcI << 5;
        __syncthreads();
        {
            float4 v = make_float4(0.f, 0.f, 0.f, 0.f);
            if (kc + k4 + 4 <= K)
                v = *(const float4*)&Amat[aRow * K + kc + k4];
            ushort4 p;
            p.x = f2bfu(v.x); p.y = f2bfu(v.y); p.z = f2bfu(v.z); p.w = f2bfu(v.w);
            *(ushort4*)sDst = p;
        }
        __syncthreads();
        v16bf af = *(const v16bf*)(&ldsA[wr][lane][0]);
        v16bf bf_ = *(const v16bf*)(Bp + (((size_t)kcI * CTpad + ct) * 32 + lane) * 16);
        acc = __builtin_amdgcn_wmma_f32_16x16x32_bf16(false, af, false, bf_, (short)0, acc, false, false);
    }
    const int col   = ct * 16 + (lane & 15);
    const int mrow0 = rowBase + wr * 16 + (lane >> 4) * 8;
    if (col < N) {
        float b = bias[col];
#pragma unroll
        for (int r = 0; r < 8; ++r)
            out[(size_t)(mrow0 + r) * N + col] = acc[r] + b;
    }
}

// ---------------------------------------------------------------------------
// small utility / elementwise kernels
// ---------------------------------------------------------------------------
__global__ void zero_buf_kernel(float* __restrict__ p, int n) {
    for (int i = blockIdx.x * blockDim.x + threadIdx.x; i < n; i += gridDim.x * blockDim.x)
        p[i] = 0.f;
}

__global__ void init_state_kernel(const float* __restrict__ h0, const float* __restrict__ h0_rel,
                                  const float* __restrict__ h0_fut, const float* __restrict__ mem_bias,
                                  const float* __restrict__ init_r,
                                  float* __restrict__ hh, float* __restrict__ hh_rel,
                                  float* __restrict__ hh_fut, float* __restrict__ mem,
                                  float* __restrict__ reads_mean)
{
    int stride = gridDim.x * blockDim.x;
    int id = blockIdx.x * blockDim.x + threadIdx.x;
    for (int i = id; i < A_ * Cc; i += stride) { hh[i] = h0[i % Cc]; hh_rel[i] = h0_rel[i % Cc]; }
    for (size_t i = id; i < (size_t)A_ * Hh * Dd; i += stride) hh_fut[i] = h0_fut[i % Dd];
    for (int i = id; i < Ss * Nn * Mm; i += stride) mem[i] = mem_bias[i % (Nn * Mm)];
    for (int i = id; i < A_ * Mm; i += stride) reads_mean[i] = init_r[i % Mm];
}

// per-step embeddings: e -> x_enc[:,0:16], reads carry -> x_enc[:,16:48], er -> er_buf
__global__ void embed_kernel(const float* __restrict__ past, const float* __restrict__ past_rel,
                             const float* __restrict__ ep_w1, const float* __restrict__ ep_b1,
                             const float* __restrict__ ep_w2, const float* __restrict__ ep_b2,
                             const float* __restrict__ er_w1, const float* __restrict__ er_b1,
                             const float* __restrict__ er_w2, const float* __restrict__ er_b2,
                             const float* __restrict__ reads_mean,
                             float* __restrict__ x_enc, float* __restrict__ er_buf, int t)
{
    int a = blockIdx.x * blockDim.x + threadIdx.x;
    if (a >= A_) return;
    float e[EMBv], er[EMBv];
    if (t < TP_) {
        float x0 = past[((size_t)a * TP_ + t) * 2], x1 = past[((size_t)a * TP_ + t) * 2 + 1];
        float y0 = past_rel[((size_t)a * TP_ + t) * 2], y1 = past_rel[((size_t)a * TP_ + t) * 2 + 1];
        float h1[8], h2[8];
#pragma unroll
        for (int j = 0; j < 8; ++j) {
            h1[j] = fmaxf(0.f, x0 * ep_w1[j] + x1 * ep_w1[8 + j] + ep_b1[j]);
            h2[j] = fmaxf(0.f, y0 * er_w1[j] + y1 * er_w1[8 + j] + er_b1[j]);
        }
#pragma unroll
        for (int j = 0; j < EMBv; ++j) {
            float s = ep_b2[j], sr = er_b2[j];
#pragma unroll
            for (int i = 0; i < 8; ++i) { s += h1[i] * ep_w2[i * EMBv + j]; sr += h2[i] * er_w2[i * EMBv + j]; }
            e[j] = s; er[j] = sr;
        }
    } else {
#pragma unroll
        for (int j = 0; j < EMBv; ++j) { e[j] = 0.f; er[j] = 0.f; }
    }
#pragma unroll
    for (int j = 0; j < EMBv; ++j) { x_enc[a * 48 + j] = e[j]; er_buf[a * EMBv + j] = er[j]; }
#pragma unroll
    for (int m = 0; m < Mm; ++m) x_enc[a * 48 + EMBv + m] = reads_mean[a * Mm + m];
}

// read attention: one block = one (agent, head); 128 threads = memory slots
__launch_bounds__(128)
__global__ void read_attn_kernel(const float* __restrict__ o, const float* __restrict__ mem,
                                 float* __restrict__ reads_h, float* __restrict__ reads_mean)
{
    int ah = blockIdx.x;
    int a = ah / Hh, s = a / Ll;
    int n = threadIdx.x;
    __shared__ float kk[Mm];
    __shared__ float sc[2];
    __shared__ float wn[Nn];
    __shared__ float red[Nn];
    const float* orow = o + (size_t)a * (Hh * 33) + (ah % Hh) * 33;
    if (n < Mm) kk[n] = orow[n];
    __syncthreads();
    if (n == 0) {
        float s2 = 0.f;
        for (int m = 0; m < Mm; ++m) s2 += kk[m] * kk[m];
        sc[0] = sqrtf(s2);
        sc[1] = softplusf_(orow[Mm]);
    }
    __syncthreads();
    const float* mrow = mem + ((size_t)s * Nn + n) * Mm;
    float dot = 0.f, m2 = 0.f;
    for (int m = 0; m < Mm; ++m) { float mv = mrow[m]; dot += kk[m] * mv; m2 += mv * mv; }
    float logit = sc[1] * (dot / (sc[0] * sqrtf(m2) + 1e-8f));
    red[n] = logit; __syncthreads();
    for (int off = 64; off > 0; off >>= 1) { if (n < off) red[n] = fmaxf(red[n], red[n + off]); __syncthreads(); }
    float mx = red[0]; __syncthreads();
    float ex = __expf(logit - mx);
    red[n] = ex; __syncthreads();
    for (int off = 64; off > 0; off >>= 1) { if (n < off) red[n] += red[n + off]; __syncthreads(); }
    wn[n] = ex / red[0];
    __syncthreads();
    if (n < Mm) {
        float r = 0.f;
        for (int q = 0; q < Nn; ++q) r += wn[q] * mem[((size_t)s * Nn + q) * Mm + n];
        reads_h[(size_t)ah * Mm + n] = r;
        atomicAdd(&reads_mean[a * Mm + n], r * (1.f / Hh));
    }
}

// write attention: one block = one agent
__launch_bounds__(128)
__global__ void write_attn_kernel(const float* __restrict__ ow, const float* __restrict__ mem,
                                  float* __restrict__ w_write, float* __restrict__ ew_buf,
                                  float* __restrict__ aw_buf)
{
    int a = blockIdx.x, s = a / Ll, n = threadIdx.x;
    __shared__ float kk[Mm];
    __shared__ float sc[2];
    __shared__ float red[Nn];
    const float* orow = ow + (size_t)a * 97;
    if (n < Mm) kk[n] = orow[n];
    __syncthreads();
    if (n == 0) {
        float s2 = 0.f;
        for (int m = 0; m < Mm; ++m) s2 += kk[m] * kk[m];
        sc[0] = sqrtf(s2);
        sc[1] = softplusf_(orow[Mm]);
    }
    __syncthreads();
    const float* mrow = mem + ((size_t)s * Nn + n) * Mm;
    float dot = 0.f, m2 = 0.f;
    for (int m = 0; m < Mm; ++m) { float mv = mrow[m]; dot += kk[m] * mv; m2 += mv * mv; }
    float logit = sc[1] * (dot / (sc[0] * sqrtf(m2) + 1e-8f));
    red[n] = logit; __syncthreads();
    for (int off = 64; off > 0; off >>= 1) { if (n < off) red[n] = fmaxf(red[n], red[n + off]); __syncthreads(); }
    float mx = red[0]; __syncthreads();
    float ex = __expf(logit - mx);
    red[n] = ex; __syncthreads();
    for (int off = 64; off > 0; off >>= 1) { if (n < off) red[n] += red[n + off]; __syncthreads(); }
    w_write[(size_t)a * Nn + n] = ex / red[0];
    if (n < Mm) {
        ew_buf[a * Mm + n] = sigmf_(orow[33 + n]);
        aw_buf[a * Mm + n] = tanhf(orow[65 + n]);
    }
}

__global__ void mem_update_kernel(const float* __restrict__ w_write, const float* __restrict__ ew,
                                  const float* __restrict__ aw, float* __restrict__ mem)
{
    int id = blockIdx.x * blockDim.x + threadIdx.x;
    if (id >= Ss * Nn * Mm) return;
    int m = id % Mm, n = (id / Mm) % Nn, s = id / (Nn * Mm);
    float keep = 1.f, add = 0.f;
#pragma unroll
    for (int l = 0; l < Ll; ++l) {
        int a = s * Ll + l;
        float ww = w_write[(size_t)a * Nn + n];
        keep *= (1.f - ww * ew[a * Mm + m]);
        add  += ww * aw[a * Mm + m];
    }
    mem[id] = mem[id] * keep + add;
}

__global__ void build_dec_in_kernel(const float* __restrict__ hh, const float* __restrict__ reads_h,
                                    float* __restrict__ dec_in)
{
    size_t total = (size_t)A_ * Hh * Dd;
    size_t stride = (size_t)gridDim.x * blockDim.x;
    for (size_t id = blockIdx.x * (size_t)blockDim.x + threadIdx.x; id < total; id += stride) {
        int i = (int)(id / Dd), k = (int)(id % Dd);
        int a = i / Hh;
        dec_in[id] = (k < Cc) ? hh[(size_t)a * Cc + k] : reads_h[(size_t)i * Mm + (k - Cc)];
    }
}

__global__ void out_proj_kernel(const float* __restrict__ hh_fut, const float* __restrict__ out_w,
                                const float* __restrict__ out_b, float* __restrict__ pred_rel, int tf)
{
    int i = blockIdx.x * blockDim.x + threadIdx.x;
    if (i >= A_ * Hh) return;
    const float* row = hh_fut + (size_t)i * Dd;
    float s0 = out_b[0], s1 = out_b[1];
    for (int k = 0; k < Dd; ++k) { float v = row[k]; s0 += v * out_w[k * 2]; s1 += v * out_w[k * 2 + 1]; }
    float* dst = pred_rel + ((size_t)i * TF_ + tf) * 2;
    dst[0] = s0; dst[1] = s1;
}

__global__ void finalize_kernel(const float* __restrict__ past, const float* __restrict__ pred_rel,
                                float* __restrict__ pred)
{
    int id = blockIdx.x * blockDim.x + threadIdx.x;
    if (id >= A_ * Hh * 2) return;
    int c = id & 1, i = id >> 1, a = i / Hh;
    float base = past[((size_t)a * TP_ + (TP_ - 1)) * 2 + c];
    float cum = 0.f;
    for (int t = 0; t < TF_; ++t) {
        cum += pred_rel[((size_t)i * TF_ + t) * 2 + c];
        pred[((size_t)i * TF_ + t) * 2 + c] = base + cum;
    }
}

// ---------------------------------------------------------------------------
extern "C" void kernel_launch(void* const* d_in, const int* in_sizes, int n_in,
                              void* d_out, int out_size, void* d_ws, size_t ws_size,
                              hipStream_t stream)
{
    (void)in_sizes; (void)n_in; (void)out_size; (void)ws_size;
    const float* past     = (const float*)d_in[0];
    const float* past_rel = (const float*)d_in[1];
    const float* ep_w1 = (const float*)d_in[2];  const float* ep_b1 = (const float*)d_in[3];
    const float* ep_w2 = (const float*)d_in[4];  const float* ep_b2 = (const float*)d_in[5];
    const float* er_w1 = (const float*)d_in[6];  const float* er_b1 = (const float*)d_in[7];
    const float* er_w2 = (const float*)d_in[8];  const float* er_b2 = (const float*)d_in[9];
    const float* enc_wih = (const float*)d_in[10]; const float* enc_whh = (const float*)d_in[11];
    const float* enc_bih = (const float*)d_in[12]; const float* enc_bhh = (const float*)d_in[13];
    const float* rel_wih = (const float*)d_in[14]; const float* rel_whh = (const float*)d_in[15];
    const float* rel_bih = (const float*)d_in[16]; const float* rel_bhh = (const float*)d_in[17];
    const float* dec_wih = (const float*)d_in[18]; const float* dec_whh = (const float*)d_in[19];
    const float* dec_bih = (const float*)d_in[20]; const float* dec_bhh = (const float*)d_in[21];
    const float* out_w = (const float*)d_in[22];  const float* out_b = (const float*)d_in[23];
    const float* fcr_w = (const float*)d_in[24];  const float* fcr_b = (const float*)d_in[25];
    const float* fcw_w = (const float*)d_in[26];  const float* fcw_b = (const float*)d_in[27];
    const float* init_r = (const float*)d_in[28];
    const float* h0     = (const float*)d_in[29];
    const float* h0_rel = (const float*)d_in[30];
    const float* h0_fut = (const float*)d_in[31];
    const float* mem_bias = (const float*)d_in[32];

    // col-tile padding (16-col tiles, groups of 4 per block)
    const int CT_C   = (Cc + 63) / 64 * 4;        // 32
    const int CT_D   = (Dd + 63) / 64 * 4;        // 36
    const int CT_O   = (Hh * 33 + 63) / 64 * 4;   // 44
    const int CT_W   = (97 + 63) / 64 * 4;        // 8
    auto nkc = [](int K) { return (K + 31) / 32; };

    // ---------------- workspace layout ----------------
    char* base = (char*)d_ws; size_t off = 0;
    auto alloc = [&](size_t bytes) -> void* {
        void* p = base + off;
        off += (bytes + 255) & ~(size_t)255;
        return p;
    };
    auto packSize = [&](int K, int ngates, int CT) { return (size_t)nkc(K) * ngates * CT * 512 * 2; };
    bf16* w_rel_ih = (bf16*)alloc(packSize(EMBv, 3, CT_C));
    bf16* w_rel_hh = (bf16*)alloc(packSize(Cc,   3, CT_C));
    bf16* w_enc_ih = (bf16*)alloc(packSize(48,   3, CT_C));
    bf16* w_enc_hh = (bf16*)alloc(packSize(Cc,   3, CT_C));
    bf16* w_dec_ih = (bf16*)alloc(packSize(Dd,   3, CT_D));
    bf16* w_dec_hh = (bf16*)alloc(packSize(Dd,   3, CT_D));
    bf16* w_fcr    = (bf16*)alloc(packSize(Cc,   1, CT_O));
    bf16* w_fcw    = (bf16*)alloc(packSize(Cc,   1, CT_W));
    float* hhA = (float*)alloc((size_t)A_ * Cc * 4);
    float* hhB = (float*)alloc((size_t)A_ * Cc * 4);
    float* hrA = (float*)alloc((size_t)A_ * Cc * 4);
    float* hrB = (float*)alloc((size_t)A_ * Cc * 4);
    float* hfA = (float*)alloc((size_t)A_ * Hh * Dd * 4);
    float* hfB = (float*)alloc((size_t)A_ * Hh * Dd * 4);
    float* x_enc  = (float*)alloc((size_t)A_ * 48 * 4);
    float* er_buf = (float*)alloc((size_t)A_ * EMBv * 4);
    float* o_buf  = (float*)alloc((size_t)A_ * Hh * 33 * 4);
    float* ow_buf = (float*)alloc((size_t)A_ * 97 * 4);
    float* reads_h    = (float*)alloc((size_t)A_ * Hh * Mm * 4);
    float* reads_mean = (float*)alloc((size_t)A_ * Mm * 4);
    float* wwr  = (float*)alloc((size_t)A_ * Nn * 4);
    float* ewb  = (float*)alloc((size_t)A_ * Mm * 4);
    float* awb  = (float*)alloc((size_t)A_ * Mm * 4);
    float* memB = (float*)alloc((size_t)Ss * Nn * Mm * 4);
    float* dec_in = (float*)alloc((size_t)A_ * Hh * Dd * 4);

    // ---------------- weight packing + state init ----------------
    auto pack = [&](const float* src, bf16* dst, int K, int gateW, int ngates, int CT) {
        pack_wmma_b_kernel<<<2048, 256, 0, stream>>>(src, dst, K, gateW, ngates, CT);
    };
    pack(rel_wih, w_rel_ih, EMBv, Cc, 3, CT_C);
    pack(rel_whh, w_rel_hh, Cc,   Cc, 3, CT_C);
    pack(enc_wih, w_enc_ih, 48,   Cc, 3, CT_C);
    pack(enc_whh, w_enc_hh, Cc,   Cc, 3, CT_C);
    pack(dec_wih, w_dec_ih, Dd,   Dd, 3, CT_D);
    pack(dec_whh, w_dec_hh, Dd,   Dd, 3, CT_D);
    pack(fcw_w,   w_fcw,    Cc,   97, 1, CT_W);
    pack_fcr_kernel<<<2048, 256, 0, stream>>>(fcr_w, w_fcr, CT_O);
    init_state_kernel<<<4096, 256, 0, stream>>>(h0, h0_rel, h0_fut, mem_bias, init_r,
                                                hhA, hrA, hfA, memB, reads_mean);

    float* hh_cur = hhA, *hh_nxt = hhB;
    float* hr_cur = hrA, *hr_nxt = hrB;
    float* hf_cur = hfA, *hf_nxt = hfB;
    float* pred = (float*)d_out;
    float* pred_rel = pred + (size_t)A_ * Hh * TF_ * 2;

    const dim3 gGru(CT_C / 4, A_ / 32);                 // 8 x 16
    const dim3 gO(CT_O / 4, A_ / 32);                   // 11 x 16
    const dim3 gW(CT_W / 4, A_ / 32);                   // 2 x 16
    const dim3 gDec(CT_D / 4, (A_ * Hh) / 32);          // 9 x 320

    for (int t = 0; t < TP_ + TF_; ++t) {
        embed_kernel<<<(A_ + 255) / 256, 256, 0, stream>>>(
            past, past_rel, ep_w1, ep_b1, ep_w2, ep_b2, er_w1, er_b1, er_w2, er_b2,
            reads_mean, x_enc, er_buf, t);
        // hh_rel = GRU(er, hh_rel)
        gru_fused_wmma<<<gGru, 256, 0, stream>>>(er_buf, EMBv, hr_cur, hr_nxt,
                                                 w_rel_ih, w_rel_hh, rel_bih, rel_bhh,
                                                 A_, Cc, CT_C, CT_C);
        // hh = GRU([e, reads], hh)
        gru_fused_wmma<<<gGru, 256, 0, stream>>>(x_enc, 48, hh_cur, hh_nxt,
                                                 w_enc_ih, w_enc_hh, enc_bih, enc_bhh,
                                                 A_, Cc, CT_C, CT_C);
        // read attention
        zero_buf_kernel<<<(A_ * Mm + 255) / 256, 256, 0, stream>>>(reads_mean, A_ * Mm);
        gemm_bias_wmma<<<gO, 256, 0, stream>>>(hh_nxt, w_fcr, fcr_b, o_buf, A_, Cc, Hh * 33, CT_O);
        read_attn_kernel<<<A_ * Hh, 128, 0, stream>>>(o_buf, memB, reads_h, reads_mean);
        // write attention + memory update
        gemm_bias_wmma<<<gW, 256, 0, stream>>>(hr_nxt, w_fcw, fcw_b, ow_buf, A_, Cc, 97, CT_W);
        write_attn_kernel<<<A_, 128, 0, stream>>>(ow_buf, memB, wwr, ewb, awb);
        mem_update_kernel<<<(Ss * Nn * Mm + 255) / 256, 256, 0, stream>>>(wwr, ewb, awb, memB);
        // decoder GRU (dominant compute)
        build_dec_in_kernel<<<4096, 256, 0, stream>>>(hh_nxt, reads_h, dec_in);
        gru_fused_wmma<<<gDec, 256, 0, stream>>>(dec_in, Dd, hf_cur, hf_nxt,
                                                 w_dec_ih, w_dec_hh, dec_bih, dec_bhh,
                                                 A_ * Hh, Dd, CT_D, CT_D);
        if (t >= TP_)
            out_proj_kernel<<<(A_ * Hh + 255) / 256, 256, 0, stream>>>(hf_nxt, out_w, out_b,
                                                                       pred_rel, t - TP_);
        float* tp;
        tp = hh_cur; hh_cur = hh_nxt; hh_nxt = tp;
        tp = hr_cur; hr_cur = hr_nxt; hr_nxt = tp;
        tp = hf_cur; hf_cur = hf_nxt; hf_nxt = tp;
    }
    finalize_kernel<<<(A_ * Hh * 2 + 255) / 256, 256, 0, stream>>>(past, pred_rel, pred);
}
